// SimpleSTFTModel_66726611911236
// MI455X (gfx1250) — matrix operand order repaired and checked by reading, not compiled, not run on hardware
//
#include <hip/hip_runtime.h>
#include <cstdint>
#include <cstddef>

typedef _Float16 half_t;
typedef __attribute__((ext_vector_type(16))) _Float16 v16h;
typedef __attribute__((ext_vector_type(8)))  float    v8f;

#define EPS_BN 1e-5f

// ============================================================ STFT ==========
// x: [128, 9000] -> out: [128][129][72]  (log2(|rfft(hann*frame)|/128 + 1e-10))
__global__ void stft_kernel(const float* __restrict__ x, float* __restrict__ out)
{
    __shared__ float fr[256];
    __shared__ float ct[256];
    __shared__ float st[256];
    int bt = blockIdx.x;            // b*72 + t
    int b = bt / 72, t = bt - b * 72;
    int k = threadIdx.x;            // 0..255
    float ph = 0.024543692606170259f * (float)k;   // 2*pi*k/256
    float cv = cosf(ph), sv = sinf(ph);
    ct[k] = cv; st[k] = sv;
    float w = 0.5f - 0.5f * cv;                    // periodic hann
    int j = t * 128 + k - 128;                     // zero boundary + tail pad
    float xv = (j >= 0 && j < 9000) ? x[b * 9000 + j] : 0.f;
    fr[k] = w * xv;
    __syncthreads();
    if (k < 129) {
        float re = 0.f, im = 0.f;
        #pragma unroll 4
        for (int n = 0; n < 256; ++n) {
            float f = fr[n];
            int m = (k * n) & 255;
            re += f * ct[m];
            im -= f * st[m];
        }
        float mag = sqrtf(re * re + im * im) * (1.0f / 128.0f); // / win.sum()
        out[(b * 129 + k) * 72 + t] = log2f(mag + 1e-10f);
    }
}

// ====================================================== conv1 (1->64) =======
// in [128][129][72] f32; fused conv3x3+bias+BN+ReLU+maxpool2x2 -> f16 [128][64][64][36]
__global__ void conv1_kernel(const float* __restrict__ in, const float* __restrict__ W,
                             const float* __restrict__ bias, const float* __restrict__ g,
                             const float* __restrict__ be, const float* __restrict__ mu,
                             const float* __restrict__ var, half_t* __restrict__ out)
{
    int idx = blockIdx.x * blockDim.x + threadIdx.x;
    if (idx >= 128 * 64 * 64 * 36) return;
    int px = idx % 36; int tmp = idx / 36;
    int py = tmp % 64; tmp /= 64;
    int c  = tmp % 64; int b = tmp / 64;
    float w9[9];
    #pragma unroll
    for (int i = 0; i < 9; ++i) w9[i] = W[c * 9 + i];
    float s  = g[c] * rsqrtf(var[c] + EPS_BN);
    float sh = bias[c] - mu[c];
    float bb = be[c];
    const float* ip = in + b * 129 * 72;
    float mx = 0.f;
    #pragma unroll
    for (int q = 0; q < 4; ++q) {
        int y = 2 * py + (q >> 1), xx = 2 * px + (q & 1);
        float acc = 0.f;
        #pragma unroll
        for (int dy = 0; dy < 3; ++dy) {
            int iy = y + dy - 1;
            if (iy < 0 || iy >= 129) continue;
            #pragma unroll
            for (int dx = 0; dx < 3; ++dx) {
                int ix = xx + dx - 1;
                if (ix < 0 || ix >= 72) continue;
                acc += w9[dy * 3 + dx] * ip[iy * 72 + ix];
            }
        }
        float v = (acc + sh) * s + bb;
        v = fmaxf(v, 0.f);
        mx = fmaxf(mx, v);
    }
    out[idx] = (half_t)mx;
}

// =========================================================== utils ==========
__global__ void fill_kernel(float* __restrict__ p, int n, float v)
{
    int i = blockIdx.x * blockDim.x + threadIdx.x;
    if (i < n) p[i] = v;
}
__global__ void pack_half_kernel(const float* __restrict__ src, half_t* __restrict__ dst, int n)
{
    int i = blockIdx.x * blockDim.x + threadIdx.x;
    if (i < n) dst[i] = (half_t)src[i];
}
// src row-major [R][C] f32 -> dst [C][R] f16   (build GEMM B matrices)
__global__ void to_half_T_kernel(const float* __restrict__ src, half_t* __restrict__ dst,
                                 int R, int C)
{
    int i = blockIdx.x * blockDim.x + threadIdx.x;
    if (i >= R * C) return;
    int r = i / C, c = i - r * C;
    dst[c * R + r] = (half_t)src[i];
}
// act5 [128][256][4][2] f32 -> seqT f16 [2048][1024]; seq[b,t,c*8+y]=act5[b,c,y/2,t/4]
__global__ void build_seq_kernel(const float* __restrict__ act5, half_t* __restrict__ seqT)
{
    int i = blockIdx.x * blockDim.x + threadIdx.x;
    if (i >= 2048 * 1024) return;
    int n = i & 1023, f = i >> 10;
    int b = n >> 3, t = n & 7;
    int c = f >> 3, y = f & 7;
    float v = act5[((b * 256 + c) * 4 + (y >> 1)) * 2 + (t >> 2)];
    seqT[i] = (half_t)v;
}

// =============================================== WMMA implicit-conv GEMM ====
// C[Co][N=B*H*W] = A[Co][Ci*9] x im2col(act); epilogue BN+ReLU+2x2 maxpool via atomicMax
template <typename T>
__global__ __launch_bounds__(128)
void conv_gemm_kernel(const half_t* __restrict__ A, const T* __restrict__ act,
                      unsigned* __restrict__ pooled,
                      int Co, int Ci, int H, int W,
                      const float* __restrict__ bias, const float* __restrict__ gam,
                      const float* __restrict__ bet, const float* __restrict__ mu,
                      const float* __restrict__ var)
{
    __shared__ __align__(16) half_t As[64 * 40];
    __shared__ __align__(16) half_t BsT[64 * 40];
    int tid = threadIdx.x;
    int nBase = blockIdx.x * 64, mBase = blockIdx.y * 64;
    int K = Ci * 9;
    int lane = tid & 31, w = tid >> 5;
    int wm = (w & 1) * 32, wn = (w >> 1) * 32;
    int kb  = (lane & 16) ? 16 : 0;     // A fragment K-base
    int kb2 = (lane & 16) ? 16 : 0;     // B fragment K-base
    int mIn = lane & 15;
    v8f acc[2][2] = {};

    // this thread's im2col column
    int nl = tid & 63;
    int col = nBase + nl;
    int xx = col % W; int tc = col / W;
    int yy = tc % H;  int bb = tc / H;
    int jh = (tid >> 6) * 16;
    const T* actb = act + (size_t)bb * Ci * H * W;

    int ai = tid >> 1, ajb = (tid & 1) * 16;

    for (int kk = 0; kk < K; kk += 32) {
        {   // A tile: 64 rows x 32 halves (Co multiple of 64 -> no guard)
            float4* dst = (float4*)&As[ai * 40 + ajb];
            const float4* s4 = (const float4*)(A + (size_t)(mBase + ai) * K + kk + ajb);
            dst[0] = s4[0]; dst[1] = s4[1];
            if (kk + 32 < K)
                __builtin_prefetch(A + (size_t)(mBase + ai) * K + kk + 32 + ajb, 0, 3);
        }
        {   // B tile (im2col on the fly), stored k-contiguous per column
            #pragma unroll
            for (int e = 0; e < 16; ++e) {
                int j = jh + e;
                int k = kk + j;
                int c = k / 9; int r9 = k - c * 9;
                int dy = r9 / 3; int dx = r9 - dy * 3;
                int iy = yy + dy - 1, ix = xx + dx - 1;
                float v = 0.f;
                if (iy >= 0 && iy < H && ix >= 0 && ix < W)
                    v = (float)actb[((size_t)c * H + iy) * W + ix];
                BsT[nl * 40 + j] = (half_t)v;
            }
        }
        __syncthreads();
        v16h bf[2];
        #pragma unroll
        for (int tj = 0; tj < 2; ++tj) {
            const half_t* pb = &BsT[(wn + tj * 16 + mIn) * 40 + kb2];
            ((float4*)&bf[tj])[0] = *(const float4*)pb;
            ((float4*)&bf[tj])[1] = *(const float4*)(pb + 8);
        }
        #pragma unroll
        for (int ti = 0; ti < 2; ++ti) {
            v16h a;
            const half_t* pa = &As[(wm + ti * 16 + mIn) * 40 + kb];
            ((float4*)&a)[0] = *(const float4*)pa;
            ((float4*)&a)[1] = *(const float4*)(pa + 8);
            #pragma unroll
            for (int tj = 0; tj < 2; ++tj)
                acc[ti][tj] = __builtin_amdgcn_wmma_f32_16x16x32_f16(
                    false, a, false, bf[tj], (short)0, acc[ti][tj], false, false);
        }
        __syncthreads();
    }

    int PH = H >> 1, PW = W >> 1;
    #pragma unroll
    for (int ti = 0; ti < 2; ++ti)
        #pragma unroll
        for (int tj = 0; tj < 2; ++tj)
            #pragma unroll
            for (int r = 0; r < 8; ++r) {
                int m = mBase + wm + ti * 16 + r + ((lane >> 4) << 3);
                int n = nBase + wn + tj * 16 + (lane & 15);
                int x2 = n % W; int t2 = n / W;
                int y2 = t2 % H; int b2 = t2 / H;
                int py = y2 >> 1, px = x2 >> 1;
                if (py < PH && px < PW) {
                    float s = gam[m] * rsqrtf(var[m] + EPS_BN);
                    float v = (acc[ti][tj][r] + bias[m] - mu[m]) * s + bet[m];
                    v = fmaxf(v, 0.f);
                    atomicMax(pooled + (((size_t)b2 * Co + m) * PH + py) * PW + px,
                              __float_as_uint(v));
                }
            }
}

// ====================================================== WMMA linear GEMM ====
// out[n*ldOut+m] = epilogue( A[M][K] x B[K][N] ), A/B f16, acc f32
__global__ __launch_bounds__(128)
void lin_gemm_kernel(const half_t* __restrict__ A, const half_t* __restrict__ Bm,
                     float* __restrict__ out, int M, int N, int K,
                     const float* __restrict__ bias0, const float* __restrict__ bias1,
                     const float* __restrict__ gam, const float* __restrict__ bet,
                     const float* __restrict__ mu, const float* __restrict__ var,
                     int relu, int Mout, int ldOut)
{
    __shared__ __align__(16) half_t As[64 * 40];
    __shared__ __align__(16) half_t BsT[64 * 40];
    int tid = threadIdx.x;
    int nBase = blockIdx.x * 64, mBase = blockIdx.y * 64;
    int lane = tid & 31, w = tid >> 5;
    int wm = (w & 1) * 32, wn = (w >> 1) * 32;
    int kb  = (lane & 16) ? 16 : 0;
    int kb2 = (lane & 16) ? 16 : 0;
    int mIn = lane & 15;
    v8f acc[2][2] = {};
    int ai = tid >> 1, ajb = (tid & 1) * 16;
    int bj = tid >> 2, bnb = (tid & 3) * 16;

    for (int kk = 0; kk < K; kk += 32) {
        {   // A tile (guard rows >= M with zeros; K multiple of 32 always)
            float4* dst = (float4*)&As[ai * 40 + ajb];
            int m = mBase + ai;
            if (m < M) {
                const float4* s4 = (const float4*)(A + (size_t)m * K + kk + ajb);
                dst[0] = s4[0]; dst[1] = s4[1];
                if (kk + 32 < K)
                    __builtin_prefetch(A + (size_t)m * K + kk + 32 + ajb, 0, 3);
            } else {
                float4 z = {0.f, 0.f, 0.f, 0.f};
                dst[0] = z; dst[1] = z;
            }
        }
        {   // B tile (N multiple of 64 always); store k-contiguous per column
            __align__(16) half_t tmp[16];
            const float4* s4 = (const float4*)(Bm + (size_t)(kk + bj) * N + nBase + bnb);
            *(float4*)&tmp[0] = s4[0];
            *(float4*)&tmp[8] = s4[1];
            #pragma unroll
            for (int e = 0; e < 16; ++e) BsT[(bnb + e) * 40 + bj] = tmp[e];
        }
        __syncthreads();
        v16h bf[2];
        #pragma unroll
        for (int tj = 0; tj < 2; ++tj) {
            const half_t* pb = &BsT[(wn + tj * 16 + mIn) * 40 + kb2];
            ((float4*)&bf[tj])[0] = *(const float4*)pb;
            ((float4*)&bf[tj])[1] = *(const float4*)(pb + 8);
        }
        #pragma unroll
        for (int ti = 0; ti < 2; ++ti) {
            v16h a;
            const half_t* pa = &As[(wm + ti * 16 + mIn) * 40 + kb];
            ((float4*)&a)[0] = *(const float4*)pa;
            ((float4*)&a)[1] = *(const float4*)(pa + 8);
            #pragma unroll
            for (int tj = 0; tj < 2; ++tj)
                acc[ti][tj] = __builtin_amdgcn_wmma_f32_16x16x32_f16(
                    false, a, false, bf[tj], (short)0, acc[ti][tj], false, false);
        }
        __syncthreads();
    }

    #pragma unroll
    for (int ti = 0; ti < 2; ++ti)
        #pragma unroll
        for (int tj = 0; tj < 2; ++tj)
            #pragma unroll
            for (int r = 0; r < 8; ++r) {
                int m = mBase + wm + ti * 16 + r + ((lane >> 4) << 3);
                int n = nBase + wn + tj * 16 + (lane & 15);
                if (m < Mout && n < N) {
                    float v = acc[ti][tj][r];
                    if (bias0) v += bias0[m];
                    if (bias1) v += bias1[m];
                    if (gam)   v = (v - mu[m]) * (gam[m] * rsqrtf(var[m] + EPS_BN)) + bet[m];
                    if (relu)  v = fmaxf(v, 0.f);
                    out[(size_t)n * ldOut + m] = v;
                }
            }
}

// ========================================================= LSTM cell ========
// gates_pre = xg[:,tIn,:] + gh ; torch gate order i,f,g,o ; H=256, B=128
__global__ void lstm_cell_kernel(const float* __restrict__ xg, int tIn,
                                 const float* __restrict__ gh, float* __restrict__ cSt,
                                 half_t* __restrict__ hT, float* __restrict__ y, int colOff,
                                 float* __restrict__ finalH, int fOff, int isLast)
{
    int idx = blockIdx.x * blockDim.x + threadIdx.x;   // < 128*256
    int b = idx >> 8, j = idx & 255;
    const float* gx = xg + (size_t)(b * 8 + tIn) * 1024;
    const float* gp = gh + (size_t)b * 1024;
    float gi = gx[j]       + gp[j];
    float gf = gx[j + 256] + gp[j + 256];
    float gg = gx[j + 512] + gp[j + 512];
    float go = gx[j + 768] + gp[j + 768];
    float ii = 1.f / (1.f + expf(-gi));
    float ff = 1.f / (1.f + expf(-gf));
    float g_ = tanhf(gg);
    float oo = 1.f / (1.f + expf(-go));
    float c = ff * cSt[idx] + ii * g_;
    cSt[idx] = c;
    float h = oo * tanhf(c);
    hT[j * 128 + b] = (half_t)h;                        // B matrix for next Whh GEMM
    if (y) y[(size_t)(b * 8 + tIn) * 512 + colOff + j] = h;
    if (isLast && finalH) finalH[(size_t)b * 512 + fOff + j] = h;
}

// ============================================================ driver ========
extern "C" void kernel_launch(void* const* d_in, const int* in_sizes, int n_in,
                              void* d_out, int out_size, void* d_ws, size_t ws_size,
                              hipStream_t stream)
{
    (void)in_sizes; (void)n_in; (void)out_size; (void)ws_size;
    auto cdiv = [](int a, int b) { return (a + b - 1) / b; };

    // ---- workspace carve-up (bytes, 256B aligned) ----
    size_t off = 0;
    auto alloc = [&](size_t bytes) {
        size_t o = off; off = (off + bytes + 255) & ~(size_t)255; return o;
    };
    char* ws = (char*)d_ws;
    float*  stftB = (float*)(ws + alloc(1188864u * 4));        // [128][129][72]
    half_t* act1  = (half_t*)(ws + alloc(18874368u * 2));      // [128][64][64][36]
    float*  act2  = (float*)(ws + alloc(9437184u * 4));        // [128][128][32][18]
    float*  act3  = (float*)(ws + alloc(4718592u * 4));        // [128][256][16][9]
    float*  act4  = (float*)(ws + alloc(2097152u * 4));        // [128][512][8][4]
    float*  act5  = (float*)(ws + alloc(262144u * 4));         // [128][256][4][2]
    half_t* wpack = (half_t*)(ws + alloc(2097152u * 2));       // reused A pack
    half_t* seqT  = (half_t*)(ws + alloc(2097152u * 2));       // [2048][1024]
    float*  xg    = (float*)(ws + alloc(1048576u * 4));        // [B*8][1024]
    float*  ghB   = (float*)(ws + alloc(131072u * 4));         // [128][1024]
    float*  cSt   = (float*)(ws + alloc(32768u * 4));
    half_t* hT    = (half_t*)(ws + alloc(32768u * 2));         // [256][128]
    float*  y1    = (float*)(ws + alloc(524288u * 4));         // [B*8][512]
    half_t* y1T   = (half_t*)(ws + alloc(524288u * 2));        // [512][1024]
    float*  zA    = (float*)(ws + alloc(131072u * 4));
    float*  zB    = (float*)(ws + alloc(131072u * 4));
    half_t* zT    = (half_t*)(ws + alloc(131072u * 2));

    const float* x = (const float*)d_in[0];

    // ---- stage 0: STFT log-magnitude ----
    stft_kernel<<<128 * 72, 256, 0, stream>>>(x, stftB);

    // ---- stage 1: conv1 fused (scalar; Cin=1) ----
    {
        const float* W  = (const float*)d_in[1];
        const float* b  = (const float*)d_in[2];
        const float* g  = (const float*)d_in[3];
        const float* be = (const float*)d_in[4];
        const float* m  = (const float*)d_in[5];
        const float* v  = (const float*)d_in[6];
        int n = 128 * 64 * 64 * 36;
        conv1_kernel<<<cdiv(n, 256), 256, 0, stream>>>(stftB, W, b, g, be, m, v, act1);
    }

    // ---- stages 2..5: implicit-GEMM WMMA convs with fused BN/ReLU/maxpool ----
    struct CC { int Ci, Co, H, W; float* out; int outN; };
    CC cc[4] = {
        {64, 128, 64, 36, act2, 9437184},
        {128, 256, 32, 18, act3, 4718592},
        {256, 512, 16, 9, act4, 2097152},
        {512, 256, 8, 4, act5, 262144},
    };
    for (int l = 0; l < 4; ++l) {
        int base = 1 + (l + 1) * 6;
        const float* W  = (const float*)d_in[base + 0];
        const float* b  = (const float*)d_in[base + 1];
        const float* g  = (const float*)d_in[base + 2];
        const float* be = (const float*)d_in[base + 3];
        const float* m  = (const float*)d_in[base + 4];
        const float* v  = (const float*)d_in[base + 5];
        int nW = cc[l].Co * cc[l].Ci * 9;
        pack_half_kernel<<<cdiv(nW, 256), 256, 0, stream>>>(W, wpack, nW);
        fill_kernel<<<cdiv(cc[l].outN, 256), 256, 0, stream>>>(cc[l].out, cc[l].outN, 0.f);
        int N = 128 * cc[l].H * cc[l].W;
        dim3 grid(N / 64, cc[l].Co / 64);
        if (l == 0)
            conv_gemm_kernel<half_t><<<grid, 128, 0, stream>>>(
                wpack, act1, (unsigned*)cc[l].out, cc[l].Co, cc[l].Ci, cc[l].H, cc[l].W,
                b, g, be, m, v);
        else
            conv_gemm_kernel<float><<<grid, 128, 0, stream>>>(
                wpack, (l == 1) ? act2 : (l == 2) ? act3 : act4, (unsigned*)cc[l].out,
                cc[l].Co, cc[l].Ci, cc[l].H, cc[l].W, b, g, be, m, v);
    }

    // ---- adaptive repeat + transpose -> seqT f16 [2048][B*8] ----
    build_seq_kernel<<<cdiv(2048 * 1024, 256), 256, 0, stream>>>(act5, seqT);

    // ---- biLSTM layers ----
    auto run_dir = [&](int dirIdx, const half_t* inT, int D,
                       float* yOut, int colOff, float* finH, int fOff) {
        const float* Wih = (const float*)d_in[31 + dirIdx * 4 + 0];
        const float* Whh = (const float*)d_in[31 + dirIdx * 4 + 1];
        const float* bih = (const float*)d_in[31 + dirIdx * 4 + 2];
        const float* bhh = (const float*)d_in[31 + dirIdx * 4 + 3];
        int nW = 1024 * D;
        pack_half_kernel<<<cdiv(nW, 256), 256, 0, stream>>>(Wih, wpack, nW);
        // xg[B*T,1024] = Wih x inT + bih + bhh
        lin_gemm_kernel<<<dim3(1024 / 64, 1024 / 64), 128, 0, stream>>>(
            wpack, inT, xg, 1024, 1024, D, bih, bhh,
            nullptr, nullptr, nullptr, nullptr, 0, 1024, 1024);
        pack_half_kernel<<<cdiv(1024 * 256, 256), 256, 0, stream>>>(Whh, wpack, 1024 * 256);
        fill_kernel<<<cdiv(32768, 256), 256, 0, stream>>>(cSt, 32768, 0.f);
        fill_kernel<<<cdiv(16384, 256), 256, 0, stream>>>((float*)hT, 16384, 0.f);
        bool bwd = (dirIdx & 1);
        for (int s = 0; s < 8; ++s) {
            int tIn = bwd ? 7 - s : s;
            // gh[B,1024] = Whh x hT
            lin_gemm_kernel<<<dim3(128 / 64, 1024 / 64), 128, 0, stream>>>(
                wpack, hT, ghB, 1024, 128, 256, nullptr, nullptr,
                nullptr, nullptr, nullptr, nullptr, 0, 1024, 1024);
            lstm_cell_kernel<<<128, 256, 0, stream>>>(
                xg, tIn, ghB, cSt, hT, yOut, colOff, finH, fOff, (s == 7) ? 1 : 0);
        }
    };

    run_dir(0, seqT, 2048, y1, 0,   nullptr, 0);   // layer1 forward
    run_dir(1, seqT, 2048, y1, 256, nullptr, 0);   // layer1 backward
    to_half_T_kernel<<<cdiv(1024 * 512, 256), 256, 0, stream>>>(y1, y1T, 1024, 512);
    run_dir(2, y1T, 512, nullptr, 0, zA, 0);       // layer2 forward -> zA[:, :256]
    run_dir(3, y1T, 512, nullptr, 0, zA, 256);     // layer2 backward -> zA[:, 256:]

    // ---- FC stack ----
    auto run_fc = [&](int fcIdx, const float* src, int Kdim, int Mdim, float* dst) {
        int base = 47 + fcIdx * 6;
        const float* W  = (const float*)d_in[base + 0];
        const float* b  = (const float*)d_in[base + 1];
        const float* g  = (const float*)d_in[base + 2];
        const float* be = (const float*)d_in[base + 3];
        const float* m  = (const float*)d_in[base + 4];
        const float* v  = (const float*)d_in[base + 5];
        to_half_T_kernel<<<cdiv(128 * Kdim, 256), 256, 0, stream>>>(src, zT, 128, Kdim);
        pack_half_kernel<<<cdiv(Mdim * Kdim, 256), 256, 0, stream>>>(W, wpack, Mdim * Kdim);
        lin_gemm_kernel<<<dim3(128 / 64, cdiv(Mdim, 64)), 128, 0, stream>>>(
            wpack, zT, dst, Mdim, 128, Kdim, b, nullptr, g, be, m, v, 1, Mdim, Mdim);
    };
    run_fc(0, zA, 512, 1024, zB);
    run_fc(1, zB, 1024, 512, zA);
    run_fc(2, zA, 512, 256, zB);

    // ---- final classifier: [4,256] x zT -> d_out [128][4] ----
    {
        const float* W = (const float*)d_in[65];
        const float* b = (const float*)d_in[66];
        to_half_T_kernel<<<cdiv(128 * 256, 256), 256, 0, stream>>>(zB, zT, 128, 256);
        pack_half_kernel<<<cdiv(4 * 256, 256), 256, 0, stream>>>(W, wpack, 4 * 256);
        lin_gemm_kernel<<<dim3(128 / 64, 1), 128, 0, stream>>>(
            wpack, zT, (float*)d_out, 4, 128, 256, b, nullptr,
            nullptr, nullptr, nullptr, nullptr, 0, 4, 4);
    }
}
